// LSTM_88776974008866
// MI455X (gfx1250) — matrix operand order repaired and checked by reading, not compiled, not run on hardware
//
#include <hip/hip_runtime.h>
#include <math.h>

// ---------------------------------------------------------------------------
// Bidirectional LSTM (B=2048, S=128, H=512, in=4) + 4 sigmoid heads.
//   * Augmented weights Waug[4H][544] bf16: [Whh | Wih | (bih+bhh) | 0pad]
//     so G_t = [h_t | x_t | 1 | 0] @ Waug^T in one WMMA K-sweep per tile.
//   * Each WG owns 16 batch rows, runs the full 128-step recurrence alone.
//   * h_aug double-buffered in LDS; x slice pre-staged in LDS; one barrier/step.
//   * Cell state c kept in persistent VGPR accumulators (per-wave tiles).
//   * v_wmma_f32_16x16x32_bf16 with ping-pong fragment double-buffering so
//     global/LDS loads for kk+1 overlap the 4 WMMAs of kk.
//   * Branchless TRANS-pipe gate math (v_exp/v_rcp/v_tanh) co-executes with
//     the matrix pipe.
// ---------------------------------------------------------------------------

typedef __bf16 bf16_t;
typedef __attribute__((ext_vector_type(16))) bf16_t v16bf;
typedef __attribute__((ext_vector_type(8)))  float  v8f;
typedef __attribute__((ext_vector_type(4)))  unsigned int u32x4;
typedef __attribute__((ext_vector_type(4)))  float  f32x4;

#define NB    2048   // batch
#define NS    128    // seq len
#define NH    512    // hidden
#define NGATE 2048   // 4*NH
#define KAUG  544    // NH + 4 (x) + 1 (bias) + 27 zero pad  (multiple of 32)
#define KITER (KAUG / 32)   // 17
#define TB    16     // batch rows per workgroup
#define NTHR  256    // 8 waves

union FragAB { v16bf v; u32x4 q[2]; };

#define LOG2E 1.4426950408889634f

__device__ __forceinline__ float fast_sig(float x) {
  return __builtin_amdgcn_rcpf(1.0f + __builtin_amdgcn_exp2f(-LOG2E * x));
}

__device__ __forceinline__ float fast_tanh(float x) {
#if __has_builtin(__builtin_amdgcn_tanhf)
  return __builtin_amdgcn_tanhf(x);
#else
  const float t = __builtin_amdgcn_exp2f((2.0f * LOG2E) * x);
  return 1.0f - 2.0f * __builtin_amdgcn_rcpf(t + 1.0f);
#endif
}

// ---------------------------------------------------------------------------
// Prep: augmented bf16 weights, row-major [NGATE][KAUG] so B fragments are
// two 16B-per-lane loads from one row.
// ---------------------------------------------------------------------------
__global__ void prep_weights_kernel(const float* __restrict__ Wih,
                                    const float* __restrict__ Whh,
                                    const float* __restrict__ bih,
                                    const float* __restrict__ bhh,
                                    bf16_t* __restrict__ Waug) {
  const int n = blockIdx.x;                       // 0..NGATE-1
  bf16_t* row = Waug + (size_t)n * KAUG;
  for (int k = threadIdx.x; k < KAUG; k += blockDim.x) {
    float v;
    if (k < NH)            v = Whh[(size_t)n * NH + k];
    else if (k < NH + 4)   v = Wih[(size_t)n * 4 + (k - NH)];
    else if (k == NH + 4)  v = bih[n] + bhh[n];
    else                   v = 0.0f;
    row[k] = (bf16_t)v;
  }
}

// ---------------------------------------------------------------------------
// Main recurrence. grid = (NB/TB, 2 directions), block = 256 (8 waves).
// ---------------------------------------------------------------------------
__global__ __launch_bounds__(NTHR)
void lstm_dir_kernel(const float* __restrict__ y,
                     const bf16_t* __restrict__ WaugF,
                     const bf16_t* __restrict__ WaugB,
                     float* __restrict__ hF,
                     float* __restrict__ hB) {
  const int rowbase = blockIdx.x * TB;
  const int dir     = blockIdx.y;                 // 0 = fwd, 1 = bwd
  const bf16_t* __restrict__ W = dir ? WaugB : WaugF;
  float* __restrict__ hout     = dir ? hB : hF;

  __shared__ bf16_t hbuf[2][TB][KAUG];            // double-buffered [h|x|1|0]
  __shared__ bf16_t xbuf[TB][NS][4];              // pre-staged inputs (16KB)

  const int tid  = threadIdx.x;
  const int wave = tid >> 5;
  const int lane = tid & 31;
  const int hl   = lane >> 4;                     // half-wave select
  const int lr   = lane & 15;

  // ---- init: zero hbuf, stage x slice (coalesced), consts, x_0 ----
  {
    unsigned int* p = (unsigned int*)&hbuf[0][0][0];
    const int nw = 2 * TB * KAUG / 2;             // 8704 dwords
    for (int i = tid; i < nw; i += NTHR) p[i] = 0u;
  }
  for (int idx = tid; idx < TB * NS * 4; idx += NTHR) {
    const int r = idx >> 9;                       // / (NS*4)
    const int rem = idx & (NS * 4 - 1);
    xbuf[r][rem >> 2][rem & 3] =
        (bf16_t)y[((size_t)(rowbase + r) * NS) * 4 + rem];
  }
  __syncthreads();
  if (tid < 2 * TB) {
    const int nb = tid >> 4, r = tid & 15;
    hbuf[nb][r][NH + 4] = (bf16_t)1.0f;
  }
  if (tid < TB * 4) {
    const int r = tid >> 2, xi = tid & 3;
    const int s0 = dir ? (NS - 1) : 0;
    hbuf[0][r][NH + xi] = xbuf[r][s0][xi];
  }
  __syncthreads();

  // persistent cell state: wave owns hidden col-tiles ct = wave*4 + c4
  v8f creg[4];
#pragma unroll
  for (int c4 = 0; c4 < 4; ++c4)
#pragma unroll
    for (int r = 0; r < 8; ++r) creg[c4][r] = 0.0f;

  for (int t = 0; t < NS; ++t) {
    const int rb = t & 1, wb = rb ^ 1;

    // stage x_{t+1} into the write buffer (LDS->LDS, disjoint regions)
    if (t + 1 < NS && tid < TB * 4) {
      const int r = tid >> 2, xi = tid & 3;
      const int s = dir ? (NS - 2 - t) : (t + 1);
      hbuf[wb][r][NH + xi] = xbuf[r][s][xi];
    }

#pragma unroll
    for (int c4 = 0; c4 < 4; ++c4) {
      const int ct = wave * 4 + c4;               // hidden col-tile 0..31
      const bf16_t* wrow0 = W + ((size_t)(0 * NH + ct * 16 + lr)) * KAUG;
      const bf16_t* wrow1 = W + ((size_t)(1 * NH + ct * 16 + lr)) * KAUG;
      const bf16_t* wrow2 = W + ((size_t)(2 * NH + ct * 16 + lr)) * KAUG;
      const bf16_t* wrow3 = W + ((size_t)(3 * NH + ct * 16 + lr)) * KAUG;

      v8f acc0 = {}, acc1 = {}, acc2 = {}, acc3 = {};

      // fragment loaders (A from LDS, B from global, ISA per-lane layouts)
      auto loadA = [&](FragAB& a, int kk) {
        const bf16_t* ap = &hbuf[rb][lr][kk * 32 + hl * 8];
        a.q[0] = *(const u32x4*)(ap);
        a.q[1] = *(const u32x4*)(ap + 16);
      };
      auto loadB = [&](FragAB& b, const bf16_t* wrow, int kk) {
        const bf16_t* bp = wrow + kk * 32 + hl * 8;
        b.q[0] = *(const u32x4*)(bp);
        b.q[1] = *(const u32x4*)(bp + 16);
      };

      // ping-pong double-buffered K loop: loads for kk+1 overlap wmmas of kk
      FragAB a0, b00, b01, b02, b03;               // even set
      FragAB a1, b10, b11, b12, b13;               // odd set

      loadA(a0, 0);
      loadB(b00, wrow0, 0); loadB(b01, wrow1, 0);
      loadB(b02, wrow2, 0); loadB(b03, wrow3, 0);

#pragma unroll 1
      for (int kp = 0; kp < (KITER - 1) / 2; ++kp) {   // 8 double-iterations
        const int kk = 2 * kp;
        loadA(a1, kk + 1);
        loadB(b10, wrow0, kk + 1); loadB(b11, wrow1, kk + 1);
        loadB(b12, wrow2, kk + 1); loadB(b13, wrow3, kk + 1);
        acc0 = __builtin_amdgcn_wmma_f32_16x16x32_bf16(false, a0.v, false, b00.v,
                                                       (short)0, acc0, false, false);
        acc1 = __builtin_amdgcn_wmma_f32_16x16x32_bf16(false, a0.v, false, b01.v,
                                                       (short)0, acc1, false, false);
        acc2 = __builtin_amdgcn_wmma_f32_16x16x32_bf16(false, a0.v, false, b02.v,
                                                       (short)0, acc2, false, false);
        acc3 = __builtin_amdgcn_wmma_f32_16x16x32_bf16(false, a0.v, false, b03.v,
                                                       (short)0, acc3, false, false);
        loadA(a0, kk + 2);
        loadB(b00, wrow0, kk + 2); loadB(b01, wrow1, kk + 2);
        loadB(b02, wrow2, kk + 2); loadB(b03, wrow3, kk + 2);
        acc0 = __builtin_amdgcn_wmma_f32_16x16x32_bf16(false, a1.v, false, b10.v,
                                                       (short)0, acc0, false, false);
        acc1 = __builtin_amdgcn_wmma_f32_16x16x32_bf16(false, a1.v, false, b11.v,
                                                       (short)0, acc1, false, false);
        acc2 = __builtin_amdgcn_wmma_f32_16x16x32_bf16(false, a1.v, false, b12.v,
                                                       (short)0, acc2, false, false);
        acc3 = __builtin_amdgcn_wmma_f32_16x16x32_bf16(false, a1.v, false, b13.v,
                                                       (short)0, acc3, false, false);
      }
      // tail: kk = 16 (loaded into the even set by the last double-iteration)
      acc0 = __builtin_amdgcn_wmma_f32_16x16x32_bf16(false, a0.v, false, b00.v,
                                                     (short)0, acc0, false, false);
      acc1 = __builtin_amdgcn_wmma_f32_16x16x32_bf16(false, a0.v, false, b01.v,
                                                     (short)0, acc1, false, false);
      acc2 = __builtin_amdgcn_wmma_f32_16x16x32_bf16(false, a0.v, false, b02.v,
                                                     (short)0, acc2, false, false);
      acc3 = __builtin_amdgcn_wmma_f32_16x16x32_bf16(false, a0.v, false, b03.v,
                                                     (short)0, acc3, false, false);

      // gate combine: c' = sig(f)*c + sig(i)*tanh(g); h = sig(o)*tanh(c')
#pragma unroll
      for (int r = 0; r < 8; ++r) {
        const int m = r + hl * 8;                 // batch row in tile
        const int j = ct * 16 + lr;               // hidden unit
        const float gi = fast_sig(acc0[r]);
        const float gf = fast_sig(acc1[r]);
        const float gg = fast_tanh(acc2[r]);
        const float go = fast_sig(acc3[r]);
        const float cn = gf * creg[c4][r] + gi * gg;
        creg[c4][r] = cn;
        hbuf[wb][m][j] = (bf16_t)(go * fast_tanh(cn));
      }
    }
    __syncthreads();   // publish h_{t+1} / x_{t+1} for next step
  }

  // ---- epilogue: final h lives in buffer 0 (wb of t = NS-1) ----
  for (int idx = tid; idx < TB * NH; idx += NTHR) {
    const int r = idx >> 9, j = idx & (NH - 1);
    hout[(size_t)(rowbase + r) * NH + j] = (float)hbuf[0][r][j];
  }
}

// ---------------------------------------------------------------------------
// Heads: out[b][o] = sigmoid(bias_o + hf[b].W_o[:512] + hb[b].W_o[512:])
// ---------------------------------------------------------------------------
__global__ void heads_kernel(const float* __restrict__ hF,
                             const float* __restrict__ hB,
                             const float* __restrict__ W1, const float* __restrict__ W2,
                             const float* __restrict__ W3, const float* __restrict__ W4,
                             const float* __restrict__ b1, const float* __restrict__ b2,
                             const float* __restrict__ b3, const float* __restrict__ b4,
                             float* __restrict__ out) {
  const int idx = blockIdx.x * blockDim.x + threadIdx.x;  // b*4 + o
  if (idx >= NB * 4) return;
  const int b = idx >> 2, o = idx & 3;
  const float* Wo = (o == 0) ? W1 : (o == 1) ? W2 : (o == 2) ? W3 : W4;
  const float bias = (o == 0) ? b1[0] : (o == 1) ? b2[0] : (o == 2) ? b3[0] : b4[0];

  const f32x4* hf = (const f32x4*)(hF + (size_t)b * NH);
  const f32x4* hb = (const f32x4*)(hB + (size_t)b * NH);
  const f32x4* wf = (const f32x4*)(Wo);
  const f32x4* wb = (const f32x4*)(Wo + NH);

  float s = bias;
  for (int k = 0; k < NH / 4; ++k) {
    const f32x4 a = hf[k], w = wf[k];
    s += a[0] * w[0] + a[1] * w[1] + a[2] * w[2] + a[3] * w[3];
  }
  for (int k = 0; k < NH / 4; ++k) {
    const f32x4 a = hb[k], w = wb[k];
    s += a[0] * w[0] + a[1] * w[1] + a[2] * w[2] + a[3] * w[3];
  }
  out[idx] = 1.0f / (1.0f + __expf(-s));
}

// ---------------------------------------------------------------------------
extern "C" void kernel_launch(void* const* d_in, const int* in_sizes, int n_in,
                              void* d_out, int out_size, void* d_ws, size_t ws_size,
                              hipStream_t stream) {
  (void)in_sizes; (void)n_in; (void)out_size; (void)ws_size;

  const float* y     = (const float*)d_in[0];
  const float* Wf_ih = (const float*)d_in[1];
  const float* Wf_hh = (const float*)d_in[2];
  const float* bf_ih = (const float*)d_in[3];
  const float* bf_hh = (const float*)d_in[4];
  const float* Wb_ih = (const float*)d_in[5];
  const float* Wb_hh = (const float*)d_in[6];
  const float* bb_ih = (const float*)d_in[7];
  const float* bb_hh = (const float*)d_in[8];
  const float* W1 = (const float*)d_in[9];  const float* b1 = (const float*)d_in[10];
  const float* W2 = (const float*)d_in[11]; const float* b2 = (const float*)d_in[12];
  const float* W3 = (const float*)d_in[13]; const float* b3 = (const float*)d_in[14];
  const float* W4 = (const float*)d_in[15]; const float* b4 = (const float*)d_in[16];

  // workspace layout
  const size_t WAUG_BYTES = (size_t)NGATE * KAUG * sizeof(bf16_t);  // 2.125 MB
  char* ws = (char*)d_ws;
  bf16_t* WaugF = (bf16_t*)(ws);
  bf16_t* WaugB = (bf16_t*)(ws + WAUG_BYTES);
  float*  hF    = (float*)(ws + 2 * WAUG_BYTES);
  float*  hB    = hF + (size_t)NB * NH;

  prep_weights_kernel<<<dim3(NGATE), 256, 0, stream>>>(Wf_ih, Wf_hh, bf_ih, bf_hh, WaugF);
  prep_weights_kernel<<<dim3(NGATE), 256, 0, stream>>>(Wb_ih, Wb_hh, bb_ih, bb_hh, WaugB);

  lstm_dir_kernel<<<dim3(NB / TB, 2), NTHR, 0, stream>>>(y, WaugF, WaugB, hF, hB);

  heads_kernel<<<dim3((NB * 4 + 255) / 256), 256, 0, stream>>>(
      hF, hB, W1, W2, W3, W4, b1, b2, b3, b4, (float*)d_out);
}
